// xLSTM_27848567947723
// MI455X (gfx1250) — matrix operand order repaired and checked by reading, not compiled, not run on hardware
//
#include <hip/hip_runtime.h>
#include <hip/hip_bf16.h>
#include <stdint.h>

// ---------------- types for CDNA5 WMMA (wave32) ----------------
typedef __bf16 bf16_t;
typedef bf16_t v16bf __attribute__((ext_vector_type(16)));
typedef float  v8f   __attribute__((ext_vector_type(8)));
typedef unsigned int v4u __attribute__((ext_vector_type(4)));
typedef int v4i_ __attribute__((vector_size(16)));   // matches async-builtin param type
typedef __attribute__((address_space(1))) v4i_* gptr4;
typedef __attribute__((address_space(3))) v4i_* lptr4;

union FragBF { v16bf v; v4u u[2]; };

// async global->LDS path (gfx1250): use only if this toolchain has the builtins
#if defined(__has_builtin)
#if __has_builtin(__builtin_amdgcn_global_load_async_to_lds_b128) && \
    __has_builtin(__builtin_amdgcn_s_wait_asynccnt)
#define USE_ASYNC_LDS 1
#endif
#endif

__device__ __forceinline__ bf16_t f2bf(float f) {
    unsigned u = __float_as_uint(f);
    unsigned r = (u + 0x7fffu + ((u >> 16) & 1u)) >> 16;   // round-to-nearest-even
    unsigned short s = (unsigned short)r;
    return __builtin_bit_cast(bf16_t, s);
}
__device__ __forceinline__ float sigmoidf_(float x) { return 1.0f / (1.0f + __expf(-x)); }

// ---------------- f32 -> bf16 convert ----------------
__global__ void __launch_bounds__(256)
cvt_f32_bf16(const float* __restrict__ src, bf16_t* __restrict__ dst, int n) {
    int i = blockIdx.x * blockDim.x + threadIdx.x;
    int stride = gridDim.x * blockDim.x;
    for (; i < n; i += stride) dst[i] = f2bf(src[i]);
}

// ---------------- time-parallel GEMM: C[M,N] = act(A[M,K] @ W[N,K]^T * scale + bias) ----
// block = 256 thr = 8 waves; WG tile 32(M) x 256(N); each wave: 16x64 via 4 wmma accums.
// Register double-buffered: k-step s+1 fragments load while k-step s WMMAs issue.
__global__ void __launch_bounds__(256)
gemm_bf16_wmma(const bf16_t* __restrict__ A, int ldA,
               const bf16_t* __restrict__ W, int ldW,
               const float* __restrict__ bias,
               float* __restrict__ Cf, bf16_t* __restrict__ Cb,
               int N, int K, float scale, int act_silu)
{
    const int lane = threadIdx.x & 31;
    const int wave = threadIdx.x >> 5;
    const int m0   = blockIdx.x * 32 + (wave >> 2) * 16;
    const int n0   = blockIdx.y * 256 + (wave & 3) * 64;
    const int half = lane >> 4;

    v8f z = {0,0,0,0,0,0,0,0};
    v8f acc[4] = { z, z, z, z };

    // A fragment (16x32 bf16, ISA 7.12.2 layout): lane holds row (lane&15),
    // K chunks [k0+half*8 .. +7] and [k0+16+half*8 .. +7]
    const bf16_t* Arow = A + (size_t)(m0 + (lane & 15)) * ldA + half * 8;
    // B fragment (32x16): lane holds col (lane&15) = W row, K chunks [k0+half*8 .. +7] and [k0+16+half*8 .. +7]
    const bf16_t* Wrow[4];
#pragma unroll
    for (int j = 0; j < 4; j++)
        Wrow[j] = W + (size_t)(n0 + j * 16 + (lane & 15)) * ldW + half * 8;

    FragBF aF[2], bF[2][4];
    auto loadK = [&](int k0, int s) {
        aF[s].u[0] = *(const v4u*)(Arow + k0);
        aF[s].u[1] = *(const v4u*)(Arow + k0 + 16);
#pragma unroll
        for (int j = 0; j < 4; j++) {
            bF[s][j].u[0] = *(const v4u*)(Wrow[j] + k0);
            bF[s][j].u[1] = *(const v4u*)(Wrow[j] + k0 + 16);
        }
    };
    auto mmaK = [&](int s) {
#pragma unroll
        for (int j = 0; j < 4; j++)
            acc[j] = __builtin_amdgcn_wmma_f32_16x16x32_bf16(
                false, aF[s].v, false, bF[s][j].v, (short)0, acc[j], false, false);
    };

    loadK(0, 0);
    for (int k0 = 0; k0 < K; k0 += 64) {            // K is a multiple of 64
        __builtin_prefetch(Arow + k0 + 256, 0, 1);  // global_prefetch_b8: stream A ahead
        loadK(k0 + 32, 1);                          // issue next loads before current math
        mmaK(0);
        if (k0 + 64 < K) loadK(k0 + 64, 0);
        mmaK(1);
    }

    const int col0 = lane & 15;
    const int rowb = m0 + half * 8;
#pragma unroll
    for (int j = 0; j < 4; j++) {
        int col = n0 + j * 16 + col0;
        float bv = bias ? bias[col] : 0.0f;
#pragma unroll
        for (int v = 0; v < 8; v++) {
            float val = acc[j][v] * scale + bv;
            if (act_silu) val = val * sigmoidf_(val);
            size_t idx = (size_t)(rowb + v) * N + col;
            if (Cf) Cf[idx] = val;
            if (Cb) Cb[idx] = f2bf(val);
        }
    }
}

// ---------------- mLSTM elementwise scan (state recurrence only) ----------------
__global__ void __launch_bounds__(256)
mlstm_scan(const float* __restrict__ Q, const float* __restrict__ Kk,
           const float* __restrict__ V, const float* __restrict__ G,
           bf16_t* __restrict__ Hb, int T, int Bn, int Hn)
{
    int e = blockIdx.x * blockDim.x + threadIdx.x;   // 0..B*H-1
    int b = e / Hn, j = e % Hn;
    float c = 0.f, nn = 0.f, mst = 0.f;
    for (int t = 0; t < T; t++) {
        size_t r = (size_t)t * Bn + b;
        size_t rn = r + Bn;                          // next timestep's row
        __builtin_prefetch(Q + rn * Hn + j, 0, 1);
        __builtin_prefetch(Kk + rn * Hn + j, 0, 1);
        __builtin_prefetch(V + rn * Hn + j, 0, 1);
        __builtin_prefetch(G + rn * (3 * Hn) + j, 0, 1);
        float q = Q[r * Hn + j], k = Kk[r * Hn + j], v = V[r * Hn + j];
        const float* g = G + r * (3 * Hn);
        float it = g[j], ft = g[Hn + j], ot = g[2 * Hn + j];
        float m2 = fmaxf(ft + mst, it);
        float i  = __expf(it - m2);
        float f  = __expf(ft + mst - m2);
        c  = f * c + i * (v * k);
        nn = f * nn + i * k;
        mst = m2;
        float h = sigmoidf_(ot) * (c * q) / fmaxf(fabsf(nn * q), 1.0f);
        Hb[r * Hn + j] = f2bf(h);
    }
}

// ---------------- sLSTM: persistent kernel, 16 WGs, 1 global barrier / step ----------
// WG wg owns H-columns [wg*32 .. wg*32+31] for ALL four gates (strided N ownership),
// so the gate update is WG-local. R fragments live in VGPRs across the whole t-loop
// (loop-invariant: 16 frags = 128 VGPRs); h double-buffered in global, staged in LDS.
__global__ void __launch_bounds__(256)
slstm_persist(const bf16_t* __restrict__ Rb,   // [2048,512] bf16 (row-major, N x K)
              const float*  __restrict__ Wx,   // [T*16, 2048] (x@W^T + b, precomputed)
              bf16_t* __restrict__ hping,      // 2 * 16*512 bf16 ping-pong
              bf16_t* __restrict__ outB,       // [T*16,512] bf16 or null
              float*  __restrict__ outF,       // [T*16,512] f32 or null
              unsigned* __restrict__ bar, int T)
{
    __shared__ bf16_t hA[16 * 512];     // A matrix (h_{t-1}) staged in LDS
    __shared__ float  pre[16][128];     // this WG's 128 preactivation columns

    const int tid  = threadIdx.x;
    const int lane = tid & 31;
    const int wave = tid >> 5;
    const int wg   = blockIdx.x;                 // 0..15
    const int j0   = wg * 32;
    // wave -> output tile: gate = wave>>1, sub = wave&1 ; n0 in [0,2048)
    const int ncol = (wave >> 1) * 512 + j0 + (wave & 1) * 16;
    const int half = lane >> 4;
    const bf16_t* Wrow = Rb + (size_t)(ncol + (lane & 15)) * 512 + half * 8;

    // ---- hoist R into registers once: loop-invariant across all T steps ----
    FragBF bw[16];
#pragma unroll
    for (int kk = 0; kk < 16; kk++) {
        bw[kk].u[0] = *(const v4u*)(Wrow + kk * 32);
        bw[kk].u[1] = *(const v4u*)(Wrow + kk * 32 + 16);
    }

    // per-thread sLSTM state for elements e = tid and tid+256 : b = e>>5, jj = e&31
    float cs[2] = {0.f, 0.f}, ns[2] = {0.f, 0.f}, ms[2] = {0.f, 0.f};

    for (int t = 0; t < T; t++) {
        // --- stage h_{t-1} (bf16) into LDS ---
        {
            const bf16_t* src = hping + (size_t)(t & 1) * 16 * 512;
#if defined(USE_ASYNC_LDS)
            for (int i = tid; i < 16 * 512 / 8; i += 256) {
                __builtin_amdgcn_global_load_async_to_lds_b128(
                    (gptr4)(src + i * 8), (lptr4)(hA + i * 8), 0, 0);
            }
            __builtin_amdgcn_s_wait_asynccnt(0);
#else
            const v4u* s4 = (const v4u*)src;
            v4u* d4 = (v4u*)hA;
            for (int i = tid; i < 16 * 512 / 8; i += 256) d4[i] = s4[i];
#endif
        }
        __syncthreads();

        // --- wave computes its 16x16 tile of h @ R^T (K = 512), R already in VGPRs ---
        v8f acc = {0,0,0,0,0,0,0,0};
        const bf16_t* Arow = hA + (lane & 15) * 512 + half * 8;
#pragma unroll
        for (int kk = 0; kk < 16; kk++) {
            FragBF a;
            a.u[0] = *(const v4u*)(Arow + kk * 32);
            a.u[1] = *(const v4u*)(Arow + kk * 32 + 16);
            acc = __builtin_amdgcn_wmma_f32_16x16x32_bf16(
                false, a.v, false, bw[kk].v, (short)0, acc, false, false);
        }

        // --- add precomputed Wx + bias, park preactivations in LDS ---
        {
            int col  = ncol + (lane & 15);
            int lcol = wave * 16 + (lane & 15);   // = gate*32 + local jj
            __builtin_prefetch(Wx + ((size_t)(t + 1) * 16 + half * 8) * 2048 + col, 0, 1);
#pragma unroll
            for (int v = 0; v < 8; v++) {
                int b = half * 8 + v;
                pre[b][lcol] = acc[v] + Wx[((size_t)t * 16 + b) * 2048 + col];
            }
        }
        __syncthreads();

        // --- WG-local gate update: 512 elements (16 b x 32 jj), 2 per thread ---
#pragma unroll
        for (int rep = 0; rep < 2; rep++) {
            int e = tid + rep * 256;
            int b = e >> 5, jj = e & 31;
            float it = pre[b][jj],      ft = pre[b][32 + jj];
            float zt = pre[b][64 + jj], ot = pre[b][96 + jj];
            float m2 = fmaxf(ft + ms[rep], it);
            float i  = __expf(it - m2);
            float f  = __expf(ft + ms[rep] - m2);
            cs[rep] = f * cs[rep] + i * tanhf(zt);
            ns[rep] = f * ns[rep] + i;
            ms[rep] = m2;
            float h = sigmoidf_(ot) * cs[rep] / ns[rep];
            int gj = j0 + jj;
            size_t ridx = ((size_t)t * 16 + b) * 512 + gj;
            if (outB) outB[ridx] = f2bf(h);
            if (outF) outF[ridx] = h;
            hping[(size_t)((t + 1) & 1) * 16 * 512 + b * 512 + gj] = f2bf(h);
        }
        __threadfence();

        // --- one global barrier per step (monotone counter, reset by init kernel) ---
        __syncthreads();
        if (tid == 0) {
            __hip_atomic_fetch_add(bar, 1u, __ATOMIC_RELEASE, __HIP_MEMORY_SCOPE_AGENT);
            unsigned target = (unsigned)(t + 1) * gridDim.x;
            while (__hip_atomic_load(bar, __ATOMIC_ACQUIRE, __HIP_MEMORY_SCOPE_AGENT) < target)
                __builtin_amdgcn_s_sleep(1);
        }
        __syncthreads();
    }
}

__global__ void __launch_bounds__(256)
slstm_init(bf16_t* hping, unsigned* bar) {
    int i = blockIdx.x * blockDim.x + threadIdx.x;
    if (i == 0) *bar = 0u;
    for (int k = i; k < 2 * 16 * 512; k += gridDim.x * blockDim.x)
        hping[k] = __builtin_bit_cast(bf16_t, (unsigned short)0);
}

// ---------------- host orchestration ----------------
extern "C" void kernel_launch(void* const* d_in, const int* in_sizes, int n_in,
                              void* d_out, int out_size, void* d_ws, size_t ws_size,
                              hipStream_t stream) {
    (void)in_sizes; (void)n_in; (void)out_size; (void)ws_size;
    constexpr int T = 2048, Bn = 16, D = 512, Hh = 512, PH = 1024;
    constexpr int M = T * Bn;                      // 32768 time-parallel rows
    const float SCALE = 0.125f;                    // 1/sqrt(H/NH)

    const float* x   = (const float*)d_in[0];
    const float* bg0 = (const float*)d_in[6];
    const float* bs1 = (const float*)d_in[9];
    const float* bg2 = (const float*)d_in[15];
    const float* bs3 = (const float*)d_in[18];

    // ---- carve scratch (bf16 weights persist; q/k/v/g region reused for Wx) ----
    char* ws = (char*)d_ws;
    size_t off = 0;
    auto carve = [&](size_t bytes) { size_t r = off; off = (off + bytes + 255) & ~(size_t)255; return r; };
    const size_t S_up = (size_t)PH * D, S_qkv = (size_t)Hh * PH,
                 S_g = (size_t)3 * Hh * PH, S_s = (size_t)4 * Hh * Hh;

    bf16_t* xb = (bf16_t*)(ws + carve((size_t)M * D * 2));
    bf16_t* wb[14];
    const size_t wsz[14] = { S_up, S_qkv, S_qkv, S_qkv, S_g, S_s, S_s,
                             S_up, S_qkv, S_qkv, S_qkv, S_g, S_s, S_s };
    const float* wsrc[14] = { (const float*)d_in[1],  (const float*)d_in[2],
                              (const float*)d_in[3],  (const float*)d_in[4],
                              (const float*)d_in[5],  (const float*)d_in[7],
                              (const float*)d_in[8],  (const float*)d_in[10],
                              (const float*)d_in[11], (const float*)d_in[12],
                              (const float*)d_in[13], (const float*)d_in[14],
                              (const float*)d_in[16], (const float*)d_in[17] };
    for (int i = 0; i < 14; i++) wb[i] = (bf16_t*)(ws + carve(wsz[i] * 2));

    bf16_t* upb = (bf16_t*)(ws + carve((size_t)M * PH * 2));
    size_t r1 = carve((size_t)M * (3 * Hh + 3 * Hh) * 4);   // q,k,v,g region; Wx aliases it
    float* qf  = (float*)(ws + r1);
    float* kf  = qf + (size_t)M * Hh;
    float* vf  = kf + (size_t)M * Hh;
    float* gf  = vf + (size_t)M * Hh;
    float* wxf = (float*)(ws + r1);                          // alias: q/k/v/g dead when live
    bf16_t* hb    = (bf16_t*)(ws + carve((size_t)M * Hh * 2));
    bf16_t* hping = (bf16_t*)(ws + carve((size_t)2 * 16 * 512 * 2));
    unsigned* bar = (unsigned*)(ws + carve(256));

    // ---- converts ----
    cvt_f32_bf16<<<256, 256, 0, stream>>>(x, xb, M * D);
    for (int i = 0; i < 14; i++)
        cvt_f32_bf16<<<256, 256, 0, stream>>>(wsrc[i], wb[i], (int)wsz[i]);

    auto gemm = [&](const bf16_t* A, int ldA, const bf16_t* W, int ldW,
                    const float* bias, float* Cf, bf16_t* Cb, int N, int K,
                    float scale, int silu) {
        gemm_bf16_wmma<<<dim3(M / 32, N / 256), 256, 0, stream>>>(
            A, ldA, W, ldW, bias, Cf, Cb, N, K, scale, silu);
    };

    // ---- layer 0: mLSTM (time-parallel projections + elementwise scan) ----
    gemm(xb, D, wb[0], D, nullptr, nullptr, upb, PH, D, 1.f, 1);          // up = silu(x@Wup^T)
    gemm(upb, PH, wb[1], PH, nullptr, qf, nullptr, Hh, PH, 1.f, 0);       // q
    gemm(upb, PH, wb[2], PH, nullptr, kf, nullptr, Hh, PH, SCALE, 0);     // k * SCALE
    gemm(upb, PH, wb[3], PH, nullptr, vf, nullptr, Hh, PH, 1.f, 0);       // v
    gemm(upb, PH, wb[4], PH, bg0, gf, nullptr, 3 * Hh, PH, 1.f, 0);       // gates
    mlstm_scan<<<32, 256, 0, stream>>>(qf, kf, vf, gf, hb, T, Bn, Hh);

    // ---- layer 1: sLSTM ----
    gemm(hb, Hh, wb[5], Hh, bs1, wxf, nullptr, 4 * Hh, Hh, 1.f, 0);       // Wx+b, all t
    slstm_init<<<1, 256, 0, stream>>>(hping, bar);
    slstm_persist<<<16, 256, 0, stream>>>(wb[6], wxf, hping, hb, nullptr, bar, T);

    // ---- layer 2: mLSTM ----
    gemm(hb, Hh, wb[7], Hh, nullptr, nullptr, upb, PH, Hh, 1.f, 1);
    gemm(upb, PH, wb[8], PH, nullptr, qf, nullptr, Hh, PH, 1.f, 0);
    gemm(upb, PH, wb[9], PH, nullptr, kf, nullptr, Hh, PH, SCALE, 0);
    gemm(upb, PH, wb[10], PH, nullptr, vf, nullptr, Hh, PH, 1.f, 0);
    gemm(upb, PH, wb[11], PH, bg2, gf, nullptr, 3 * Hh, PH, 1.f, 0);
    mlstm_scan<<<32, 256, 0, stream>>>(qf, kf, vf, gf, hb, T, Bn, Hh);

    // ---- layer 3: sLSTM (writes final f32 output) ----
    gemm(hb, Hh, wb[12], Hh, bs3, wxf, nullptr, 4 * Hh, Hh, 1.f, 0);
    slstm_init<<<1, 256, 0, stream>>>(hping, bar);
    slstm_persist<<<16, 256, 0, stream>>>(wb[13], wxf, hping, nullptr, (float*)d_out, bar, T);
}
